// RGAT_65266323030535
// MI455X (gfx1250) — matrix operand order repaired and checked by reading, not compiled, not run on hardware
//
#include <hip/hip_runtime.h>
#include <hip/hip_bf16.h>

// ---------------------------------------------------------------------------
// RGCN (2-layer) for MI455X / gfx1250, wave32 + WMMA bf16.
//
// Layer:  out = relu( [A0/c0 | ... | A7/c7 | x] @ [W0;...;W7;root] + b )
// where A[r][dst] = sum over edges of relation r into dst of x[src].
//
// Aggregate-first dataflow: x (25.6 MB) is L2-resident during the edge pass;
// the dense 50000 x 1152 x 128 GEMM runs on v_wmma_f32_16x16x32_bf16 with
// f32 accumulation. Memory-bound (~0.7 GB/layer @ 23.3 TB/s).
// ---------------------------------------------------------------------------

typedef __attribute__((ext_vector_type(16))) __bf16 v16bf;
typedef __attribute__((ext_vector_type(8)))  __bf16 v8bf;
typedef __attribute__((ext_vector_type(8)))  float  v8f;

#define N_NODES 50000
#define N_EDGES 500000
#define HIDDEN  128
#define N_REL   8
#define KBLKS   9           // 8 relations + root

__device__ __forceinline__ __bf16 f2bf(float f) {
    union { float f; unsigned u; } v; v.f = f;
    unsigned r = (v.u + 0x7FFFu + ((v.u >> 16) & 1u)) >> 16;   // RNE
    union { unsigned short s; __bf16 b; } o; o.s = (unsigned short)r;
    return o.b;
}

// ---------------------------------------------------------------------------
// Pack [W (8x128x128, k-major) ; root (128x128, k-major)] -> bf16, transposed
// to wT[blk][out][k] so each lane's B-fragment is two contiguous 16B loads.
// ---------------------------------------------------------------------------
__global__ void convert_w(const float* __restrict__ W,
                          const float* __restrict__ root,
                          __bf16* __restrict__ wT) {
    int idx = blockIdx.x * blockDim.x + threadIdx.x;      // over 9*128*128
    if (idx >= KBLKS * HIDDEN * HIDDEN) return;
    int k   = idx & (HIDDEN - 1);
    int o   = (idx >> 7) & (HIDDEN - 1);
    int blk = idx >> 14;
    float v = (blk < N_REL) ? W[((size_t)blk * HIDDEN + k) * HIDDEN + o]
                            : root[(size_t)k * HIDDEN + o];
    wT[idx] = f2bf(v);
}

// ---------------------------------------------------------------------------
// One wave32 per edge: lane j adds x[src][4j..4j+3] into A[rel][dst][...],
// lane 0 bumps cnt[dst*8+rel].
// ---------------------------------------------------------------------------
__global__ __launch_bounds__(256)
void aggregate(const float* __restrict__ xin,
               const int*   __restrict__ ei,   // [2][E]
               const int*   __restrict__ et,   // [E]
               float* __restrict__ A,          // [8][N][128]
               float* __restrict__ cnt) {      // [N][8]
    int gtid = blockIdx.x * blockDim.x + threadIdx.x;
    int e    = gtid >> 5;
    int lane = gtid & 31;
    if (e >= N_EDGES) return;
    int s = ei[e];
    int d = ei[N_EDGES + e];
    int r = et[e];

    const float4 v = *(const float4*)(xin + (size_t)s * HIDDEN + lane * 4);
    float* dst = A + (((size_t)r * N_NODES + d) * HIDDEN) + lane * 4;
    atomicAdd(dst + 0, v.x);
    atomicAdd(dst + 1, v.y);
    atomicAdd(dst + 2, v.z);
    atomicAdd(dst + 3, v.w);
    if (lane == 0) atomicAdd(&cnt[(size_t)d * N_REL + r], 1.0f);
}

// ---------------------------------------------------------------------------
// 256 threads = 8 waves per block; block computes out[16 rows][128 cols].
// Wave w owns columns [16w, 16w+16). K = 9 blocks of 128; each block's
// normalized bf16 panel is double-buffered in LDS so staging loads overlap
// the 4x v_wmma_f32_16x16x32_bf16 of the previous block; 1 barrier/iter.
// ---------------------------------------------------------------------------
#define SA_PITCH 136   // 16B-aligned rows, 4-bank skew per row -> conflict free

__global__ __launch_bounds__(256)
void rgcn_gemm(const float*  __restrict__ A,     // [8][N][128] sums
               const float*  __restrict__ cnt,   // [N][8]
               const float*  __restrict__ xin,   // [N][128]
               const __bf16* __restrict__ wT,    // [9][128(out)][128(k)] bf16
               const float*  __restrict__ bias,  // [128]
               float*        __restrict__ out) { // [N][128]
    __shared__ __bf16 sA[2][16][SA_PITCH];

    const int tid     = threadIdx.x;
    const int lane    = tid & 31;
    const int wave    = tid >> 5;              // 0..7 -> 16-col slice
    const int halfsel = lane >> 4;             // 0 | 1 (lane group)
    const int row0    = blockIdx.x * 16;
    const int n       = (wave << 4) | (lane & 15);   // output column

    // staging coords: each thread loads 8 floats of one row
    const int sr   = tid >> 4;                 // 0..15 (row in tile)
    const int sc   = (tid & 15) * 8;           // 0..120 (col)
    const int node = row0 + sr;

    auto stage = [&](int blk, int buf) {
        if (blk < N_REL) {
            float c  = cnt[(size_t)node * N_REL + blk];
            float ic = __builtin_amdgcn_rcpf(fmaxf(c, 1.0f));  // v_rcp_f32
            const float* src = A + (((size_t)blk * N_NODES + node) * HIDDEN) + sc;
#pragma unroll
            for (int j = 0; j < 8; ++j) sA[buf][sr][sc + j] = f2bf(src[j] * ic);
        } else {
            const float* src = xin + (size_t)node * HIDDEN + sc;
#pragma unroll
            for (int j = 0; j < 8; ++j) sA[buf][sr][sc + j] = f2bf(src[j]);
        }
    };

    v8f acc;
    const float bv = bias[n];
#pragma unroll
    for (int i = 0; i < 8; ++i) acc[i] = bv;

    stage(0, 0);
    __syncthreads();

    for (int blk = 0; blk < KBLKS; ++blk) {
        const int buf = blk & 1;
        // prefetch next panel into the other buffer (overlaps WMMA below)
        if (blk + 1 < KBLKS) stage(blk + 1, buf ^ 1);

        const __bf16* wb = wT + ((size_t)blk * HIDDEN + n) * HIDDEN;
#pragma unroll
        for (int ks = 0; ks < 4; ++ks) {
            const int kc = ks * 32;
            // A fragment: lane (m = lane&15) holds K kc+g..kc+g+7 and +16
            const __bf16* ap = &sA[buf][lane & 15][kc + halfsel * 8];
            v8bf alo = *(const v8bf*)ap;
            v8bf ahi = *(const v8bf*)(ap + 16);
            // B fragment: lane (col n) holds K kc+g..kc+g+7 and +16
            const __bf16* bp = wb + kc + halfsel * 8;
            v8bf blo = *(const v8bf*)bp;
            v8bf bhi = *(const v8bf*)(bp + 16);
            v16bf af, bf;
#pragma unroll
            for (int i = 0; i < 8; ++i) {
                af[i] = alo[i]; af[8 + i] = ahi[i];
                bf[i] = blo[i]; bf[8 + i] = bhi[i];
            }
            acc = __builtin_amdgcn_wmma_f32_16x16x32_bf16(
                false, af, false, bf, (short)0, acc, false, false);
        }
        __syncthreads();
    }

    // ---- relu + store: VGPR i -> row (i + 8*halfsel) ----
    const int m0 = halfsel * 8;
#pragma unroll
    for (int i = 0; i < 8; ++i) {
        float v = acc[i];
        v = v > 0.0f ? v : 0.0f;
        out[(size_t)(row0 + m0 + i) * HIDDEN + n] = v;
    }
}

// ---------------------------------------------------------------------------

extern "C" void kernel_launch(void* const* d_in, const int* in_sizes, int n_in,
                              void* d_out, int out_size, void* d_ws, size_t ws_size,
                              hipStream_t stream) {
    const float* x    = (const float*)d_in[0];
    const int*   ei   = (const int*)  d_in[1];
    const int*   et   = (const int*)  d_in[2];
    /* d_in[3] = edge_attr, unused by the reference math */
    const float* W1   = (const float*)d_in[4];
    const float* r1   = (const float*)d_in[5];
    const float* b1   = (const float*)d_in[6];
    const float* W2   = (const float*)d_in[7];
    const float* r2   = (const float*)d_in[8];
    const float* b2   = (const float*)d_in[9];
    float* outp = (float*)d_out;

    // workspace partition
    char* ws = (char*)d_ws;
    const size_t A_bytes   = (size_t)N_REL * N_NODES * HIDDEN * sizeof(float); // 204.8 MB
    const size_t cnt_bytes = (size_t)N_NODES * N_REL * sizeof(float);          // 1.6 MB
    const size_t h_bytes   = (size_t)N_NODES * HIDDEN * sizeof(float);         // 25.6 MB
    float*  Abuf = (float*)ws;                        ws += A_bytes;
    float*  cnt  = (float*)ws;                        ws += cnt_bytes;
    float*  h    = (float*)ws;                        ws += h_bytes;
    __bf16* wT   = (__bf16*)ws;

    const int conv_grid = (KBLKS * HIDDEN * HIDDEN + 255) / 256;   // 576
    const int agg_grid  = (N_EDGES * 32 + 255) / 256;              // 62500
    const int gemm_grid = N_NODES / 16;                            // 3125

    // ---------------- layer 1 ----------------
    convert_w<<<conv_grid, 256, 0, stream>>>(W1, r1, wT);
    hipMemsetAsync(Abuf, 0, A_bytes + cnt_bytes, stream);  // A + cnt contiguous
    aggregate<<<agg_grid, 256, 0, stream>>>(x, ei, et, Abuf, cnt);
    rgcn_gemm<<<gemm_grid, 256, 0, stream>>>(Abuf, cnt, x, wT, b1, h);

    // ---------------- layer 2 ----------------
    convert_w<<<conv_grid, 256, 0, stream>>>(W2, r2, wT);
    hipMemsetAsync(Abuf, 0, A_bytes + cnt_bytes, stream);
    aggregate<<<agg_grid, 256, 0, stream>>>(h, ei, et, Abuf, cnt);
    rgcn_gemm<<<gemm_grid, 256, 0, stream>>>(Abuf, cnt, h, wT, b2, outp);
}
